// Linear4bit_3736621548095
// MI455X (gfx1250) — 8-bit matrix operand order repaired and checked by reading, compiled, not run on hardware
//
#include <hip/hip_runtime.h>

typedef int v8i __attribute__((ext_vector_type(8)));
typedef unsigned int u32;

namespace {
constexpr int MDIM   = 8192;
constexpr int NDIM   = 8192;
constexpr int KDIM   = 8192;
constexpr int KBYTE  = KDIM / 2;
constexpr int TILE   = 128;
constexpr int KSTEPB = 32;
constexpr int PITCH  = 80;
}

__device__ __forceinline__ void unpack_xor(u32 v, u32& e, u32& o) {
  e = (v & 0x0F0F0F0Fu) ^ 0x08080808u;
  o = ((v >> 4) & 0x0F0F0F0Fu) ^ 0x08080808u;
}

__device__ __forceinline__ u32 pack_low_bytes(int4 pv) {
#if __has_builtin(__builtin_amdgcn_perm)
  u32 t0 = __builtin_amdgcn_perm((u32)pv.y, (u32)pv.x, 0x00000400u);
  u32 t1 = __builtin_amdgcn_perm((u32)pv.w, (u32)pv.z, 0x00000400u);
  return __builtin_amdgcn_perm(t1, t0, 0x05040100u);
#else
  return (u32)(pv.x & 0xFF) | ((u32)(pv.y & 0xFF) << 8) |
         ((u32)(pv.z & 0xFF) << 16) | ((u32)(pv.w & 0xFF) << 24);
#endif
}

__global__ __launch_bounds__(256) void i4_rowsum_kernel(const int* __restrict__ x,
                                                        int* __restrict__ rs) {
  const int wave = threadIdx.x >> 5;
  const int lane = threadIdx.x & 31;
  const int row  = blockIdx.x * 8 + wave;
  const int* p = x + (size_t)row * KDIM;
  int s = 0;
  for (int k = lane; k < KBYTE; k += 32) {
    u32 b = (u32)p[k] & 0xFFu;
    s += (int)(((b & 0xFu) ^ 8u) + (((b >> 4) & 0xFu) ^ 8u));
  }
  for (int off = 16; off; off >>= 1) s += __shfl_xor(s, off, 32);
  if (lane == 0) rs[row] = s;
}

__global__ __launch_bounds__(256) void i4_colsum_kernel(const u32* __restrict__ w,
                                                        int* __restrict__ cs) {
  const int wave = threadIdx.x >> 5;
  const int lane = threadIdx.x & 31;
  const int row  = blockIdx.x * 8 + wave;
  const u32* p = w + (size_t)row * (KBYTE / 4);
  u32 acc2 = 0;
  for (int k = lane; k < KBYTE / 4; k += 32) {
    u32 v = p[k];
    u32 e, o;
    unpack_xor(v, e, o);
    u32 s8 = e + o;
    acc2 += (s8 & 0x00FF00FFu) + ((s8 >> 8) & 0x00FF00FFu);
  }
  int s = (int)((acc2 & 0xFFFFu) + (acc2 >> 16));
  for (int off = 16; off; off >>= 1) s += __shfl_xor(s, off, 32);
  if (lane == 0) cs[row] = s;
}

__global__ __launch_bounds__(256) void i4_gemm_wmma_kernel(
    const int* __restrict__ x,
    const unsigned char* __restrict__ wq,
    const _Float16* __restrict__ bias,
    const int* __restrict__ rowsum,
    const int* __restrict__ colsum,
    float* __restrict__ out)
{
  __shared__ __align__(16) unsigned char As[2][TILE][PITCH];
  __shared__ __align__(16) unsigned char Bs[2][TILE][PITCH];

  const int tid   = threadIdx.x;
  const int lane  = tid & 31;
  const int wave  = tid >> 5;
  const int wm    = wave >> 2;
  const int wn    = wave & 3;
  const int li    = lane & 15;
  const int half  = lane >> 4;
  const int tileM = blockIdx.y * TILE;
  const int tileN = blockIdx.x * TILE;

  const int lrow  = tid >> 1;
  const int lhalf = tid & 1;

  v8i acc[4][2];
#pragma unroll
  for (int mi = 0; mi < 4; ++mi)
#pragma unroll
    for (int ni = 0; ni < 2; ++ni) acc[mi][ni] = 0;

  u32 a_up[8];
  u32 b_up[8];

  const int ksteps = KBYTE / KSTEPB;

  auto fetch = [&](int s) {
    const int kb = s * KSTEPB + lhalf * 16;
    const int* gx = x + (size_t)(tileM + lrow) * KDIM + kb;
#pragma unroll
    for (int j = 0; j < 4; ++j) {
      int4 pv = ((const int4*)gx)[j];
      u32 pk = pack_low_bytes(pv);
      unpack_xor(pk, a_up[2 * j], a_up[2 * j + 1]);
    }
    const unsigned char* gw = wq + (size_t)(tileN + lrow) * KBYTE + kb;
    uint4 bv = *(const uint4*)gw;
    unpack_xor(bv.x, b_up[0], b_up[1]);
    unpack_xor(bv.y, b_up[2], b_up[3]);
    unpack_xor(bv.z, b_up[4], b_up[5]);
    unpack_xor(bv.w, b_up[6], b_up[7]);
    if (s + 2 < ksteps) {
      __builtin_prefetch(gx + 2 * KSTEPB, 0, 3);
      __builtin_prefetch(gw + 2 * KSTEPB, 0, 3);
    }
  };

  auto commit = [&](int buf) {
    uint4* pa = (uint4*)&As[buf][lrow][lhalf * 32];
    pa[0] = make_uint4(a_up[0], a_up[1], a_up[2], a_up[3]);
    pa[1] = make_uint4(a_up[4], a_up[5], a_up[6], a_up[7]);
    uint4* pb = (uint4*)&Bs[buf][lrow][lhalf * 32];
    pb[0] = make_uint4(b_up[0], b_up[1], b_up[2], b_up[3]);
    pb[1] = make_uint4(b_up[4], b_up[5], b_up[6], b_up[7]);
  };

  auto compute = [&](int buf) {
    v8i bf[2];
#pragma unroll
    for (int ni = 0; ni < 2; ++ni) {
      const unsigned char* bp = &Bs[buf][wn * 32 + ni * 16 + li][0];
      int4 u0 = *(const int4*)(bp + half * 16);
      int4 u1 = *(const int4*)(bp + 32 + half * 16);
      bf[ni][0] = u0.x; bf[ni][1] = u0.y; bf[ni][2] = u0.z; bf[ni][3] = u0.w;
      bf[ni][4] = u1.x; bf[ni][5] = u1.y; bf[ni][6] = u1.z; bf[ni][7] = u1.w;
    }
#pragma unroll
    for (int mi = 0; mi < 4; ++mi) {
      const unsigned char* ap = &As[buf][wm * 64 + mi * 16 + li][0];
      v8i af;
#pragma unroll
      for (int q = 0; q < 4; ++q) {
        int2 d = *(const int2*)(ap + (q >> 1) * 32 + half * 16 + (q & 1) * 8);
        af[2 * q]     = d.x;
        af[2 * q + 1] = d.y;
      }
#pragma unroll
      for (int ni = 0; ni < 2; ++ni) {
        acc[mi][ni] = __builtin_amdgcn_wmma_i32_16x16x64_iu8(
            false, af, false, bf[ni], acc[mi][ni], false, false);
      }
    }
  };

  fetch(0);
  commit(0);
  __syncthreads();
  for (int s = 0; s < ksteps; ++s) {
    const int buf = s & 1;
    if (s + 1 < ksteps) fetch(s + 1);
    compute(buf);
    if (s + 1 < ksteps) {
      commit(buf ^ 1);
      __syncthreads();
    }
  }

  const int rbase = tileM + wm * 64;
  const int cbase = tileN + wn * 32;
#pragma unroll
  for (int ni = 0; ni < 2; ++ni) {
    const int col = cbase + ni * 16 + li;
    const _Float16 bv = bias[col];
    const int cs = colsum[col];
#pragma unroll
    for (int mi = 0; mi < 4; ++mi) {
#pragma unroll
      for (int r = 0; r < 8; ++r) {
        const int row = rbase + mi * 16 + half * 8 + r;
        const int t = acc[mi][ni][r] - 8 * (rowsum[row] + cs) + 64 * KDIM;
        _Float16 hv = (_Float16)(float)t + bv;
        out[(size_t)row * NDIM + col] = (float)hv;
      }
    }
  }
}

extern "C" void kernel_launch(void* const* d_in, const int* in_sizes, int n_in,
                              void* d_out, int out_size, void* d_ws, size_t ws_size,
                              hipStream_t stream) {
  (void)in_sizes; (void)n_in; (void)out_size; (void)ws_size;
  const int* x             = (const int*)d_in[0];
  const unsigned char* wq  = (const unsigned char*)d_in[1];
  const _Float16* bias     = (const _Float16*)d_in[2];
  float* out               = (float*)d_out;
  int* rowsum              = (int*)d_ws;
  int* colsum              = rowsum + MDIM;

  i4_rowsum_kernel<<<MDIM / 8, 256, 0, stream>>>(x, rowsum);
  i4_colsum_kernel<<<NDIM / 8, 256, 0, stream>>>((const u32*)wq, colsum);

  dim3 grid(NDIM / TILE, MDIM / TILE);
  i4_gemm_wmma_kernel<<<grid, 256, 0, stream>>>(x, wq, bias, rowsum, colsum, out);
}
